// LearnForces_36971078484285
// MI455X (gfx1250) — matrix operand order repaired and checked by reading, not compiled, not run on hardware
//
#include <hip/hip_runtime.h>
#include <math.h>

typedef __attribute__((ext_vector_type(16))) _Float16 v16h;
typedef __attribute__((ext_vector_type(8)))  float    v8f;

#define NP 16
#define NE 120
#define HID 128
#define HSTRIDE 136   // halfs per activation row (68 dwords -> conflict-free A loads)
#define WSTRIDE 69    // dwords per packed-weight column row (stride%64=5 -> conflict-free B loads)

union AF { v16h v; unsigned int u[8]; };
union P2 { _Float16 h[2]; unsigned int u; };

// One WMMA MLP layer: hout = relu(hin @ W + b), 128x128 @ 128x128, f16 in / f32 acc.
// hin: [128][HSTRIDE] f16 row-major.  wp: packed B-fragments wp[col][k2] = {W[2k2][col], W[2k2+1][col]}.
__device__ __forceinline__ void mlp_layer(const _Float16 (*hin)[HSTRIDE],
                                          _Float16 (*hout)[HSTRIDE],
                                          const unsigned int (*wp)[WSTRIDE],
                                          const float* __restrict__ bias,
                                          int tid)
{
    const int wav = tid >> 5;
    const int lane = tid & 31;
    const int ln = lane & 15;     // laneM for A, laneN for B/C/D
    const int hi = lane >> 4;

    // A fragments for this wave's 16-row M-tile, all four K=32 steps.
    // ISA layout (16-bit A 16x32): dword r holds K = 16*(r>>2) + 2*(r&3) + 8*hi (+kbase)
    AF a[4];
    const unsigned int* rowp = (const unsigned int*)&hin[wav * 16 + ln][0];
#pragma unroll
    for (int k4 = 0; k4 < 4; ++k4) {
#pragma unroll
        for (int r = 0; r < 8; ++r)
            a[k4].u[r] = rowp[16 * k4 + (r >> 2) * 8 + (r & 3) + hi * 4];
    }

#pragma unroll
    for (int n = 0; n < 8; ++n) {
        // bias depends only on N (C/D layout: N = lane&15 for every acc VGPR)
        float bv = bias[n * 16 + ln];
        v8f acc;
#pragma unroll
        for (int v = 0; v < 8; ++v) acc[v] = bv;

        const unsigned int* col = &wp[n * 16 + ln][0];
#pragma unroll
        for (int k4 = 0; k4 < 4; ++k4) {
            // ISA layout (16-bit B 32x16): dword r holds K = kbase + 16*(r>>2) + 2*(r&3) + 8*hi  ->  k2 = 16*k4 + 8*(r>>2) + (r&3) + 4*hi
            AF b;
#pragma unroll
            for (int r = 0; r < 8; ++r)
                b.u[r] = col[k4 * 16 + (r >> 2) * 8 + (r & 3) + hi * 4];
            acc = __builtin_amdgcn_wmma_f32_16x16x32_f16(
                false, a[k4].v, false, b.v, (short)0, acc, false, false);
        }
        // relu + f16 store; D layout: M = v + 8*hi (+16*wav), N = ln (+16*n)
        // (ternary == v_max_num(x,0): NaN -> 0, no canonicalize op)
#pragma unroll
        for (int v = 0; v < 8; ++v) {
            float rv = acc[v] > 0.0f ? acc[v] : 0.0f;
            hout[wav * 16 + v + 8 * hi][n * 16 + ln] = (_Float16)rv;
        }
    }
}

__global__ __launch_bounds__(256) void
forces_kernel(const float* __restrict__ D,
              const float* __restrict__ logm,
              const float* __restrict__ logG,
              const float* __restrict__ W1, const float* __restrict__ b1,
              const float* __restrict__ W2, const float* __restrict__ b2,
              const float* __restrict__ W3, const float* __restrict__ b3,
              const float* __restrict__ W4, const float* __restrict__ b4,
              const int* __restrict__ send, const int* __restrict__ recv,
              float* __restrict__ out)
{
    __shared__ _Float16 hbuf[2][HID][HSTRIDE];       // ping-pong activations (f16)
    __shared__ unsigned int wpk[2][HID][WSTRIDE];    // W2/W3 packed into B-fragment order
    __shared__ float sfeat[HID][8];
    __shared__ float sW1[6][HID];
    __shared__ float sb1[HID], sb2[HID], sb3[HID];
    __shared__ float sW4[HID][4];
    __shared__ float sb4[4];
    __shared__ float lmass[NP];
    __shared__ float nodef[NP][4];
    __shared__ int ssend[NE], srecv[NE];

    const int tid = threadIdx.x;
    const int t = blockIdx.x;   // one time step per block

    // ---------- phase 0: stage params into LDS ----------
    for (int i = tid; i < 6 * HID; i += 256) sW1[i / HID][i % HID] = W1[i];
    if (tid < HID) { sb1[tid] = b1[tid]; sb2[tid] = b2[tid]; sb3[tid] = b3[tid]; }
    for (int i = tid; i < HID * 3; i += 256) sW4[i / 3][i % 3] = W4[i];
    if (tid < 3) sb4[tid] = b4[tid];
    if (tid == 0) lmass[0] = 0.74212575f;             // log10(M_sun)
    else if (tid < NP) lmass[tid] = logm[tid - 1];
    if (tid < NE) { ssend[tid] = send[tid]; srecv[tid] = recv[tid]; }
    if (tid < NP * 4) ((float*)nodef)[tid] = 0.0f;

    // pack W2/W3 -> f16 B-fragment layout: wpk[l][col][k2] = {W[2k2][col], W[2k2+1][col]}
    for (int i = tid; i < HID * 64; i += 256) {
        int n = i & 127, k2 = i >> 7;
        P2 p;
        p.h[0] = (_Float16)W2[(2 * k2) * HID + n];
        p.h[1] = (_Float16)W2[(2 * k2 + 1) * HID + n];
        wpk[0][n][k2] = p.u;
        P2 q;
        q.h[0] = (_Float16)W3[(2 * k2) * HID + n];
        q.h[1] = (_Float16)W3[(2 * k2 + 1) * HID + n];
        wpk[1][n][k2] = q.u;
    }
    __syncthreads();

    // ---------- phase 1: edge features (cart2sph + masses + logG) ----------
    const float gG = logG[0];
    if (tid < NE) {
        int e = t * NE + tid;
        float x = D[3 * e], y = D[3 * e + 1], z = D[3 * e + 2];
        float r = sqrtf(x * x + y * y + z * z);
        float zn = z / fmaxf(r, 1e-12f);
        zn = fminf(fmaxf(zn, -1.0f), 1.0f);
        sfeat[tid][0] = r;
        sfeat[tid][1] = acosf(zn);
        sfeat[tid][2] = atan2f(y, x);
        sfeat[tid][3] = lmass[srecv[tid]];
        sfeat[tid][4] = lmass[ssend[tid]];
        sfeat[tid][5] = gG;
        sfeat[tid][6] = 0.0f; sfeat[tid][7] = 0.0f;
    } else if (tid < HID) {
#pragma unroll
        for (int c = 0; c < 8; ++c) sfeat[tid][c] = 0.0f;  // pad rows 120..127
    }
    __syncthreads();

    // ---------- layer 1 (6 -> 128), scalar: 384 FMA/thread ----------
    {
        int c = tid & 127;
        int r0 = (tid >> 7) * 64;
        float wc0 = sW1[0][c], wc1 = sW1[1][c], wc2 = sW1[2][c];
        float wc3 = sW1[3][c], wc4 = sW1[4][c], wc5 = sW1[5][c];
        float bc = sb1[c];
        for (int j = r0; j < r0 + 64; ++j) {
            float acc = bc;
            acc += sfeat[j][0] * wc0; acc += sfeat[j][1] * wc1;
            acc += sfeat[j][2] * wc2; acc += sfeat[j][3] * wc3;
            acc += sfeat[j][4] * wc4; acc += sfeat[j][5] * wc5;
            float rv = acc > 0.0f ? acc : 0.0f;
            hbuf[0][j][c] = (_Float16)rv;
        }
    }
    __syncthreads();

    // ---------- layer 2 (128 -> 128) via WMMA ----------
    mlp_layer(hbuf[0], hbuf[1], wpk[0], sb2, tid);
    __syncthreads();

    // ---------- layer 3 (128 -> 128) via WMMA ----------
    mlp_layer(hbuf[1], hbuf[0], wpk[1], sb3, tid);
    __syncthreads();

    // ---------- layer 4 (128 -> 3) + sph2cart + local scatter ----------
    if (tid < NE) {
        float e0 = sb4[0], e1 = sb4[1], e2 = sb4[2];
        const unsigned int* rp = (const unsigned int*)&hbuf[0][tid][0];
#pragma unroll 4
        for (int k2 = 0; k2 < 64; ++k2) {
            P2 p; p.u = rp[k2];                       // two halves per ds_load_b32
            float h0 = (float)p.h[0];
            float h1 = (float)p.h[1];
            int k = 2 * k2;
            e0 += h0 * sW4[k][0] + h1 * sW4[k + 1][0];
            e1 += h0 * sW4[k][1] + h1 * sW4[k + 1][1];
            e2 += h0 * sW4[k][2] + h1 * sW4[k + 1][2];
        }
        float st = sinf(e1), ct = cosf(e1);
        float sp = sinf(e2), cp = cosf(e2);
        float fx = e0 * st * cp, fy = e0 * st * sp, fz = e0 * ct;
        int s = ssend[tid], r = srecv[tid];
        atomicAdd(&nodef[r][0], fx);  atomicAdd(&nodef[r][1], fy);  atomicAdd(&nodef[r][2], fz);
        atomicAdd(&nodef[s][0], -fx); atomicAdd(&nodef[s][1], -fy); atomicAdd(&nodef[s][2], -fz);
    }
    __syncthreads();

    // ---------- output: accel = (f_recv - f_sent) / 10^lm ----------
    if (tid < NP * 3) {
        int p = tid / 3, c = tid % 3;
        out[(t * NP + p) * 3 + c] = nodef[p][c] * exp10f(-lmass[p]);
    }
}

extern "C" void kernel_launch(void* const* d_in, const int* in_sizes, int n_in,
                              void* d_out, int out_size, void* d_ws, size_t ws_size,
                              hipStream_t stream) {
    const float* D     = (const float*)d_in[0];
    const float* logm  = (const float*)d_in[1];
    const float* logG  = (const float*)d_in[2];
    const float* W1    = (const float*)d_in[3];
    const float* b1    = (const float*)d_in[4];
    const float* W2    = (const float*)d_in[5];
    const float* b2    = (const float*)d_in[6];
    const float* W3    = (const float*)d_in[7];
    const float* b3    = (const float*)d_in[8];
    const float* W4    = (const float*)d_in[9];
    const float* b4    = (const float*)d_in[10];
    const int*   send  = (const int*)d_in[11];
    const int*   recv  = (const int*)d_in[12];
    float* out = (float*)d_out;

    int ntime = in_sizes[0] / (3 * NE);   // 8192
    forces_kernel<<<ntime, 256, 0, stream>>>(D, logm, logG, W1, b1, W2, b2,
                                             W3, b3, W4, b4, send, recv, out);
}
